// VisualTokenSelection_6150393168245
// MI455X (gfx1250) — matrix operand order repaired and checked by reading, not compiled, not run on hardware
//
#include <hip/hip_runtime.h>
#include <hip/hip_bf16.h>

typedef __attribute__((ext_vector_type(16))) _Float16 v16h;
typedef __attribute__((ext_vector_type(8)))  _Float16 v8h;
typedef __attribute__((ext_vector_type(8)))  float    v8f;
typedef __attribute__((ext_vector_type(4)))  unsigned v4u;
typedef __attribute__((ext_vector_type(8)))  int      v8i;
typedef __attribute__((ext_vector_type(4)))  int      v4i;

#define NTOK   50
#define MPAD   64
#define DIM    512
#define CH     256
#define HD     128
#define NSEL   48
#define TOPK_N 12
#define NSAMP  500
#define SIGMA  0.05f
#define NB     192   // 16 * MAX_FRAMES

__device__ __forceinline__ float gelu_exact(float v) {
  return 0.5f * v * (1.0f + erff(v * 0.7071067811865475f));
}

__device__ __forceinline__ unsigned hashu(unsigned x) {
  x ^= x >> 16; x *= 0x7feb352dU;
  x ^= x >> 15; x *= 0x846ca68bU;
  x ^= x >> 16;
  return x;
}

// ---------------------------------------------------------------------------
// K-loop accumulator for one 16x16 output tile, f16 A (LDS row-major) and
// f16 Bt (N-major / K-contiguous, LDS or global), f32 accumulate via WMMA.
// A layout: lanes 0-15 = M rows, lanes 0-15 hold K..K+15, lanes 16-31 hold
// K+16..K+31 (contiguous, matching B).
// B layout: lanes 0-15 hold K..K+15, lanes 16-31 hold K+16..K+31 (contiguous).
// ---------------------------------------------------------------------------
__device__ __forceinline__ v8f wmma_accum(const _Float16* __restrict__ arow,
                                          const _Float16* __restrict__ brow,
                                          int K, v8f acc, int half) {
  #pragma unroll 4
  for (int k = 0; k < K; k += 32) {
    v16h a  = *(const v16h*)(arow + k + half * 16);
    v16h bm = *(const v16h*)(brow + k + half * 16);
    acc = __builtin_amdgcn_wmma_f32_16x16x32_f16(false, a, false, bm, (short)0, acc,
                                                 false, false);
  }
  return acc;
}

// STORE: 0=f16, 1=f16+gelu, 2=f32, 3=f16 transposed (D[n][m]), 4=f16+bias+gelu
template <int STORE>
__device__ __forceinline__ void gemm(const _Float16* __restrict__ A, int lda,
                                     const _Float16* __restrict__ Bt, int ldb,
                                     void* Dv, int ldd,
                                     int m16, int n16, int K,
                                     const float* __restrict__ bias) {
  const int tid  = threadIdx.x;
  const int lane = tid & 31;
  const int wave = tid >> 5;
  const int half = lane >> 4;
  const int l15  = lane & 15;
  const int ntiles = m16 * n16;
  for (int t = wave; t < ntiles; t += 8) {
    const int mt = t / n16, nt = t - mt * n16;
    const int mB = mt * 16, nB = nt * 16;
    const _Float16* arow = A  + (mB + l15) * lda;
    const _Float16* brow = Bt + (nB + l15) * ldb;
    v8f acc = {};
    acc = wmma_accum(arow, brow, K, acc, half);
    const int n = nB + l15;
    #pragma unroll
    for (int r = 0; r < 8; ++r) {
      const int m = mB + r + half * 8;
      float v = acc[r];
      if (STORE == 1) v = gelu_exact(v);
      if (STORE == 4) v = gelu_exact(v + bias[n]);
      if (STORE == 2)      ((float*)Dv)[m * ldd + n]     = v;
      else if (STORE == 3) ((_Float16*)Dv)[n * ldd + m]  = (_Float16)v;
      else                 ((_Float16*)Dv)[m * ldd + n]  = (_Float16)v;
    }
  }
}

// ---------------------------------------------------------------------------
// Weight prep: f32 (K,N) row-major -> f16 transposed (N,K) K-contiguous.
// ---------------------------------------------------------------------------
__global__ void txp_kernel(const float* __restrict__ src, _Float16* __restrict__ dst,
                           int K, int N) {
  int i = blockIdx.x * blockDim.x + threadIdx.x;
  if (i >= K * N) return;
  int n = i / K, k = i - n * K;
  dst[i] = (_Float16)src[k * N + n];
}

// ---------------------------------------------------------------------------
// Fused predictor: LN -> gelu(x@w_in) -> MHA -> split-concat MLP -> tanh score
// One block (8 wave32) per frame-batch b; all activations live in LDS.
// w2 is staged LDS-side via the Tensor Data Mover (TDM) where available.
// ---------------------------------------------------------------------------
__global__ __launch_bounds__(256)
void predictor_kernel(const float* __restrict__ x,
                      const float* __restrict__ ln_g, const float* __restrict__ ln_b,
                      const _Float16* __restrict__ w_inT,
                      const _Float16* __restrict__ wqT, const _Float16* __restrict__ wkT,
                      const _Float16* __restrict__ wvT, const _Float16* __restrict__ woT,
                      const _Float16* __restrict__ w1T, const float* __restrict__ w2,
                      float* __restrict__ scoresOut) {
  __shared__ alignas(32) _Float16 sA[MPAD * DIM];    // 64KB: ln_x; then Q|K; then proj
  __shared__ alignas(32) _Float16 sH[MPAD * CH];     // 32KB: h; then O; then h4
  __shared__ alignas(32) _Float16 sVT[CH * MPAD];    // 32KB: v transposed [dk][tok]
  __shared__ alignas(32) float    sS[MPAD * MPAD];   // 16KB: attention scores
  __shared__ alignas(32) _Float16 sP[MPAD * MPAD];   // 8KB : softmax probs
  __shared__ float sG[CH];
  __shared__ float sR[CH];
  __shared__ alignas(16) float sW2[CH];              // 1KB : TDM-staged w2

  const int b   = blockIdx.x;
  const int tid = threadIdx.x;
  const float* xb = x + (size_t)b * NTOK * DIM;

  // Warm the L2-resident weight streams.
  __builtin_prefetch(w_inT + tid * 64, 0, 1);
  __builtin_prefetch(w1T   + tid * 64, 0, 1);

#if defined(__gfx1250__) && __has_builtin(__builtin_amdgcn_tensor_load_to_lds)
  // ---- TDM: DMA w2 (256 x f32, 2D tile 256x1) global -> LDS ----
  bool tdm_wave = (tid < 32);
  if (tdm_wave) {
    // LDS aperture: low 32 bits of the generic pointer are the LDS offset.
    unsigned lds_off = (unsigned)(unsigned long long)(const void*)sW2;
    unsigned long long ga = (unsigned long long)(const void*)w2;
    // D# group0: [1:0]=count=1, [63:32]=lds_addr, [120:64]=global_addr, [127:126]=type 2
    v4u g0 = { 1u, lds_off,
               (unsigned)(ga & 0xffffffffu),
               (unsigned)((ga >> 32) & 0x01ffffffu) | (2u << 30) };
    // D# group1: data_size=2(4B) @[17:16]; tensor_dim0=256 @[79:48];
    //            tensor_dim1=1 @[111:80]; tile_dim0=256 @[127:112];
    //            tile_dim1=1 @[143:128]; tensor_dim0_stride=256 @[207:160]
    v8i g1 = { (int)(2u << 16),
               (int)((256u & 0xffffu) << 16),
               (int)(1u << 16),
               (int)(256u << 16),
               (int)1,
               (int)256,
               0, 0 };
    v4i gz4 = { 0, 0, 0, 0 };
    v8i gz8 = { 0, 0, 0, 0, 0, 0, 0, 0 };
    __builtin_amdgcn_tensor_load_to_lds(g0, g1, gz4, gz4, gz8, 0);
  }
#else
  if (tid < CH) sW2[tid] = w2[tid];
#endif

  // ---- load x tile -> f16 LDS, zero-pad rows 50..63 ----
  for (int i = tid; i < MPAD * DIM; i += 256) {
    int m = i >> 9, c = i & (DIM - 1);
    float v = (m < NTOK) ? xb[m * DIM + c] : 0.0f;
    sA[i] = (_Float16)v;
  }
  __syncthreads();

  // ---- LayerNorm in place (one thread per row) ----
  if (tid < MPAD) {
    float mu = 0.f;
    for (int c = 0; c < DIM; ++c) mu += (float)sA[tid * DIM + c];
    mu *= (1.0f / DIM);
    float var = 0.f;
    for (int c = 0; c < DIM; ++c) { float d = (float)sA[tid * DIM + c] - mu; var += d * d; }
    var *= (1.0f / DIM);
    float inv = rsqrtf(var + 1e-5f);
    for (int c = 0; c < DIM; ++c) {
      float v = ((float)sA[tid * DIM + c] - mu) * inv * ln_g[c] + ln_b[c];
      sA[tid * DIM + c] = (_Float16)v;
    }
  }
  __syncthreads();

  // ---- h = gelu(ln_x @ w_in)  (64x512)@(512x256) ----
  gemm<1>(sA, DIM, w_inT, DIM, sH, CH, 4, 16, DIM, nullptr);
  __syncthreads();

  // ---- q, k, v (v stored transposed) ----
  _Float16* sQ = sA;
  _Float16* sK = sA + MPAD * CH;
  gemm<0>(sH, CH, wqT, CH, sQ, CH, 4, 16, CH, nullptr);
  gemm<0>(sH, CH, wkT, CH, sK, CH, 4, 16, CH, nullptr);
  gemm<3>(sH, CH, wvT, CH, sVT, MPAD, 4, 16, CH, nullptr);
  __syncthreads();

  // ---- attention per head; O overwrites sH ----
  for (int hh = 0; hh < 2; ++hh) {
    // S = q_h @ k_h^T : k row-major IS the N-major/K-contiguous B operand
    gemm<2>(sQ + hh * HD, CH, sK + hh * HD, CH, sS, MPAD, 4, 4, HD, nullptr);
    __syncthreads();
    if (tid < MPAD) {
      const float scale = 0.08838834764831845f;  // 1/sqrt(128)
      float mx = -3.4e38f;
      for (int n = 0; n < NTOK; ++n) mx = fmaxf(mx, sS[tid * MPAD + n] * scale);
      float sum = 0.f;
      for (int n = 0; n < NTOK; ++n) {
        float e = __expf(sS[tid * MPAD + n] * scale - mx);
        sS[tid * MPAD + n] = e; sum += e;
      }
      float inv = 1.0f / sum;
      for (int n = 0; n < MPAD; ++n)
        sP[tid * MPAD + n] = (_Float16)((n < NTOK) ? sS[tid * MPAD + n] * inv : 0.0f);
    }
    __syncthreads();
    // O_h = P @ v_h : vT is already the N-major B operand
    gemm<0>(sP, MPAD, sVT + hh * HD * MPAD, MPAD, sH + hh * HD, CH, 4, 8, MPAD, nullptr);
    __syncthreads();
  }

  // ---- proj = O @ wo -> first half of sA ----
  _Float16* sProj = sA;
  gemm<0>(sH, CH, woT, CH, sProj, CH, 4, 16, CH, nullptr);
  __syncthreads();

  // ---- pooled mean g, then rank-1 bias row r = g @ w1[256:,:] ----
  if (tid < CH) {
    float s = 0.f;
    for (int t = 0; t < NTOK; ++t) s += (float)sProj[t * CH + tid];
    sG[tid] = s * (1.0f / NTOK);
  }
  __syncthreads();
  if (tid < CH) {
    float s = 0.f;
    const _Float16* wrow = w1T + tid * DIM + CH;  // w1T[n][256..511]
    for (int c = 0; c < CH; ++c) s += sG[c] * (float)wrow[c];
    sR[tid] = s;
  }
  __syncthreads();

  // ---- h4 = gelu(proj @ w1[:256,:] + r) -> sH ----
  gemm<4>(sProj, CH, w1T, DIM, sH, CH, 4, 16, CH, sR);
  __syncthreads();

#if defined(__gfx1250__) && __has_builtin(__builtin_amdgcn_tensor_load_to_lds)
  // Ensure the TDM transfer landed before reading sW2.
  if (tid < 32) __builtin_amdgcn_s_wait_tensorcnt(0);
  __syncthreads();
#endif

  // ---- pred = tanh(h4 @ w2); emit spatial scores [2:50) ----
  if (tid < MPAD) {
    float s = 0.f;
    for (int c = 0; c < CH; ++c) s += (float)sH[tid * CH + c] * sW2[c];
    float p = tanhf(s);
    if (tid >= 2 && tid < NTOK) scoresOut[b * NSEL + (tid - 2)] = p;
  }
}

// ---------------------------------------------------------------------------
// Perturbed top-k + token selection + cls copy. One block per frame-batch.
// ---------------------------------------------------------------------------
__global__ __launch_bounds__(256)
void select_kernel(const float* __restrict__ x, const float* __restrict__ scores,
                   float* __restrict__ out) {
  __shared__ float sSc[NSEL];
  __shared__ float sCnt[TOPK_N * NSEL];
  const int b = blockIdx.x;
  const int tid = threadIdx.x;

  if (tid < NSEL) sSc[tid] = scores[b * NSEL + tid];
  for (int i = tid; i < TOPK_N * NSEL; i += 256) sCnt[i] = 0.f;
  __syncthreads();

  for (int s = tid; s < NSAMP; s += 256) {
    float val[NSEL];
    const unsigned base = ((unsigned)(b * NSAMP + s)) * 64u + 0x9e3779b9U;
    #pragma unroll
    for (int p = 0; p < NSEL / 2; ++p) {
      unsigned h1 = hashu(base + 2u * p);
      unsigned h2 = hashu(base + 2u * p + 1u);
      float u1 = ((float)(h1 >> 8) + 0.5f) * (1.0f / 16777216.0f);
      float u2 = (float)(h2 >> 8) * (1.0f / 16777216.0f);
      float rr = sqrtf(-2.0f * __logf(u1));
      float th = 6.28318530717958647f * u2;
      val[2 * p]     = sSc[2 * p]     + SIGMA * rr * __cosf(th);
      val[2 * p + 1] = sSc[2 * p + 1] + SIGMA * rr * __sinf(th);
    }
    // top-12 via 12 masked argmax rounds over a register-resident array
    unsigned long long selm = 0ull;
    for (int r = 0; r < TOPK_N; ++r) {
      float best = -3.4e38f; int bi = 0;
      #pragma unroll
      for (int l = 0; l < NSEL; ++l) {
        bool freebit = ((selm >> l) & 1ull) == 0ull;
        if (freebit && val[l] > best) { best = val[l]; bi = l; }
      }
      selm |= (1ull << bi);
    }
    // streaming the mask low->high yields index-sorted ranks directly
    int rank = 0;
    #pragma unroll
    for (int l = 0; l < NSEL; ++l) {
      if ((selm >> l) & 1ull) {
        atomicAdd(&sCnt[rank * NSEL + l], 1.0f);
        ++rank;
      }
    }
  }
  __syncthreads();

  // sel = (cnt/500) @ spatial_feat ; plus cls rows 0,1 copied from x
  const float invS = 1.0f / (float)NSAMP;
  const float* xb = x + (size_t)b * NTOK * DIM;
  float* ob = out + (size_t)b * 14 * DIM;
  for (int d = tid; d < DIM; d += 256) {
    ob[0 * DIM + d] = xb[0 * DIM + d];
    ob[1 * DIM + d] = xb[1 * DIM + d];
    float acc[TOPK_N];
    #pragma unroll
    for (int k = 0; k < TOPK_N; ++k) acc[k] = 0.f;
    for (int l = 0; l < NSEL; ++l) {
      float xv = xb[(2 + l) * DIM + d];
      #pragma unroll
      for (int k = 0; k < TOPK_N; ++k) acc[k] += sCnt[k * NSEL + l] * xv;
    }
    #pragma unroll
    for (int k = 0; k < TOPK_N; ++k) ob[(2 + k) * DIM + d] = acc[k] * invS;
  }
}

// ---------------------------------------------------------------------------
extern "C" void kernel_launch(void* const* d_in, const int* in_sizes, int n_in,
                              void* d_out, int out_size, void* d_ws, size_t ws_size,
                              hipStream_t stream) {
  (void)in_sizes; (void)n_in; (void)out_size; (void)ws_size;
  const float* x    = (const float*)d_in[0];
  const float* ln_g = (const float*)d_in[1];
  const float* ln_b = (const float*)d_in[2];
  const float* w_in = (const float*)d_in[3];
  const float* wq   = (const float*)d_in[4];
  const float* wk   = (const float*)d_in[5];
  const float* wv   = (const float*)d_in[6];
  const float* wo   = (const float*)d_in[7];
  const float* w1   = (const float*)d_in[8];
  const float* w2   = (const float*)d_in[9];

  char* ws = (char*)d_ws;
  _Float16* w_inT = (_Float16*)(ws);                       // 256x512 f16
  _Float16* wqT   = (_Float16*)(ws + 262144);              // 256x256
  _Float16* wkT   = (_Float16*)(ws + 262144 + 131072);
  _Float16* wvT   = (_Float16*)(ws + 262144 + 2 * 131072);
  _Float16* woT   = (_Float16*)(ws + 262144 + 3 * 131072);
  _Float16* w1T   = (_Float16*)(ws + 262144 + 4 * 131072); // 256x512
  float*    sc    = (float*)   (ws + 2 * 262144 + 4 * 131072); // 192x48 f32

  txp_kernel<<<(512 * 256 + 255) / 256, 256, 0, stream>>>(w_in, w_inT, 512, 256);
  txp_kernel<<<(256 * 256 + 255) / 256, 256, 0, stream>>>(wq, wqT, 256, 256);
  txp_kernel<<<(256 * 256 + 255) / 256, 256, 0, stream>>>(wk, wkT, 256, 256);
  txp_kernel<<<(256 * 256 + 255) / 256, 256, 0, stream>>>(wv, wvT, 256, 256);
  txp_kernel<<<(256 * 256 + 255) / 256, 256, 0, stream>>>(wo, woT, 256, 256);
  txp_kernel<<<(512 * 256 + 255) / 256, 256, 0, stream>>>(w1, w1T, 512, 256);

  predictor_kernel<<<NB, 256, 0, stream>>>(x, ln_g, ln_b, w_inT, wqT, wkT, wvT, woT,
                                           w1T, w2, sc);
  select_kernel<<<NB, 256, 0, stream>>>(x, sc, (float*)d_out);
}
